// RNNLayer_85907935855144
// MI455X (gfx1250) — matrix operand order repaired and checked by reading, not compiled, not run on hardware
//
#include <hip/hip_runtime.h>

typedef __attribute__((ext_vector_type(16))) __bf16 v16bf;
typedef __attribute__((ext_vector_type(8)))  float  v8f;

#define T_STEPS 128
#define N_ENV   512
#define IN_DIMC 512
#define HID     512
#define H3      1536
#define LN_EPS  1e-5f

// ---------------------------------------------------------------- helpers

__device__ __forceinline__ float sigmoid_fast(float x) {
    return 1.0f / (1.0f + __expf(-x));
}

__device__ __forceinline__ float tanh_fast(float x) {
    float a = fabsf(x);
    float e = __expf(-2.0f * a);
    float t = (1.0f - e) / (1.0f + e);
    return copysignf(t, x);
}

__device__ __forceinline__ v8f wmma_bf16(v16bf a, v16bf b, v8f c) {
    // D = A(16x32 bf16) x B(32x16 bf16) + C(16x16 f32)
    return __builtin_amdgcn_wmma_f32_16x16x32_bf16(
        /*neg_a=*/false, a, /*neg_b=*/false, b,
        /*c_mod=*/(short)0, c, /*reuse_a=*/false, /*reuse_b=*/false);
}

// A-fragment (16x32 bf16, this lane = row M=lane%16):
//   element e <- K = k0 + 16*halfsel + e
// p must already point at  row*pitch + k0 + 16*halfsel  (16B aligned).
__device__ __forceinline__ v16bf load_a_frag(const float* __restrict__ p, float scale) {
    const float4* q = (const float4*)p;
    float4 q0 = q[0], q1 = q[1];   // K offsets 0..7
    float4 q2 = q[2], q3 = q[3];   // K offsets 8..15
    v16bf a;
    a[0]  = (__bf16)(q0.x * scale); a[1]  = (__bf16)(q0.y * scale);
    a[2]  = (__bf16)(q0.z * scale); a[3]  = (__bf16)(q0.w * scale);
    a[4]  = (__bf16)(q1.x * scale); a[5]  = (__bf16)(q1.y * scale);
    a[6]  = (__bf16)(q1.z * scale); a[7]  = (__bf16)(q1.w * scale);
    a[8]  = (__bf16)(q2.x * scale); a[9]  = (__bf16)(q2.y * scale);
    a[10] = (__bf16)(q2.z * scale); a[11] = (__bf16)(q2.w * scale);
    a[12] = (__bf16)(q3.x * scale); a[13] = (__bf16)(q3.y * scale);
    a[14] = (__bf16)(q3.z * scale); a[15] = (__bf16)(q3.w * scale);
    return a;
}

// B-fragment (32x16 bf16, this lane = col N=lane%16): element e <- K = k0 + 16*halfsel + e.
// p points at W[(c)*pitch + k0 + 16*halfsel]  (32B aligned) -> single vector load.
__device__ __forceinline__ v16bf load_b_frag(const __bf16* __restrict__ p) {
    return *(const v16bf*)p;
}

// ---------------------------------------------------------------- kernels

// Convert fp32 weights (3H x 512 each) to bf16 in workspace (done once per call).
__global__ void cvt_weights_kernel(const float* __restrict__ Wih,
                                   const float* __restrict__ Whh,
                                   __bf16* __restrict__ Wih_bf,
                                   __bf16* __restrict__ Whh_bf) {
    int idx = blockIdx.x * blockDim.x + threadIdx.x;
    if (idx < H3 * IN_DIMC) {
        Wih_bf[idx] = (__bf16)Wih[idx];
        Whh_bf[idx] = (__bf16)Whh[idx];
    }
}

// One GRU timestep, fully fused: gi = x_t@Wih^T, gh = (h*mask)@Whh^T, gates, h_new.
// Wave = 32(M) x 16(N) block of h_new; 12 independent bf16 WMMA accumulation chains.
__global__ void __launch_bounds__(128)
gru_step_kernel(const float* __restrict__ x_t,     // (N_ENV, IN_DIMC)
                const float* __restrict__ h_prev,  // (N_ENV, HID) un-masked
                const float* __restrict__ mask_t,  // (N_ENV)
                const __bf16* __restrict__ Wih,    // (3H, IN_DIMC) bf16
                const __bf16* __restrict__ Whh,    // (3H, HID) bf16
                const float* __restrict__ b_ih,    // (3H)
                const float* __restrict__ b_hh,    // (3H)
                float* __restrict__ h_out,         // (N_ENV, HID)
                float* __restrict__ h_last,        // (N_ENV, HID) tail of d_out
                int write_last) {
    const int lane    = threadIdx.x & 31;
    const int wave    = threadIdx.x >> 5;
    const int wid     = blockIdx.x * 4 + wave;  // 0..511
    const int mb      = wid >> 5;               // 0..15 : 32-row block
    const int nt      = wid & 31;               // 0..31 : 16-col tile
    const int m0      = mb * 32;
    const int j       = nt * 16;
    const int halfsel = lane >> 4;
    const int lrow    = lane & 15;

    const v8f vzero = {0.f, 0.f, 0.f, 0.f, 0.f, 0.f, 0.f, 0.f};
    v8f acc_gi[3][2], acc_gh[3][2];
#pragma unroll
    for (int g = 0; g < 3; ++g)
#pragma unroll
        for (int mi = 0; mi < 2; ++mi) { acc_gi[g][mi] = vzero; acc_gh[g][mi] = vzero; }

    // per-lane A-row pointers (A-matrix row M = lane%16) and reset masks
    const float mk0 = mask_t[m0 + lrow];
    const float mk1 = mask_t[m0 + 16 + lrow];
    const float* xrow0 = x_t   + (size_t)(m0 + lrow)      * IN_DIMC + 16 * halfsel;
    const float* xrow1 = x_t   + (size_t)(m0 + 16 + lrow) * IN_DIMC + 16 * halfsel;
    const float* hrow0 = h_prev + (size_t)(m0 + lrow)      * HID     + 16 * halfsel;
    const float* hrow1 = h_prev + (size_t)(m0 + 16 + lrow) * HID     + 16 * halfsel;

    // per-lane B-col pointers (B-matrix col N = lane%16); gate bases r/z/n
    const __bf16* bihp[3] = {
        Wih + (size_t)(       j + lrow) * IN_DIMC + 16 * halfsel,
        Wih + (size_t)( 512 + j + lrow) * IN_DIMC + 16 * halfsel,
        Wih + (size_t)(1024 + j + lrow) * IN_DIMC + 16 * halfsel };
    const __bf16* bhhp[3] = {
        Whh + (size_t)(       j + lrow) * HID + 16 * halfsel,
        Whh + (size_t)( 512 + j + lrow) * HID + 16 * halfsel,
        Whh + (size_t)(1024 + j + lrow) * HID + 16 * halfsel };

    for (int k0 = 0; k0 < IN_DIMC; k0 += 32) {
        v16bf ax0 = load_a_frag(xrow0 + k0, 1.0f);
        v16bf ax1 = load_a_frag(xrow1 + k0, 1.0f);
        v16bf ah0 = load_a_frag(hrow0 + k0, mk0);
        v16bf ah1 = load_a_frag(hrow1 + k0, mk1);
#pragma unroll
        for (int g = 0; g < 3; ++g) {
            v16bf bi = load_b_frag(bihp[g] + k0);
            acc_gi[g][0] = wmma_bf16(ax0, bi, acc_gi[g][0]);
            acc_gi[g][1] = wmma_bf16(ax1, bi, acc_gi[g][1]);
            v16bf bh = load_b_frag(bhhp[g] + k0);
            acc_gh[g][0] = wmma_bf16(ah0, bh, acc_gh[g][0]);
            acc_gh[g][1] = wmma_bf16(ah1, bh, acc_gh[g][1]);
        }
    }

    // fused gate epilogue; C/D layout: element i -> (M = i + 8*halfsel, N = lane%16)
    const int   col = j + lrow;
    const float bir = b_ih[col], biz = b_ih[512 + col], bin = b_ih[1024 + col];
    const float bhr = b_hh[col], bhz = b_hh[512 + col], bhn = b_hh[1024 + col];
#pragma unroll
    for (int mi = 0; mi < 2; ++mi) {
#pragma unroll
        for (int i = 0; i < 8; ++i) {
            const int row = m0 + mi * 16 + i + 8 * halfsel;
            const float hm = h_prev[(size_t)row * HID + col] * mask_t[row];
            const float r  = sigmoid_fast(acc_gi[0][mi][i] + bir + acc_gh[0][mi][i] + bhr);
            const float z  = sigmoid_fast(acc_gi[1][mi][i] + biz + acc_gh[1][mi][i] + bhz);
            const float nn = tanh_fast(acc_gi[2][mi][i] + bin + r * (acc_gh[2][mi][i] + bhn));
            const float hn = (1.0f - z) * nn + z * hm;
            h_out[(size_t)row * HID + col] = hn;
            if (write_last) h_last[(size_t)row * HID + col] = hn;
        }
    }
}

// In-place LayerNorm over each row of (T*N, HID); one wave32 per row.
__global__ void __launch_bounds__(256)
ln_kernel(float* __restrict__ out,
          const float* __restrict__ ln_w,
          const float* __restrict__ ln_b) {
    const int lane = threadIdx.x & 31;
    const int row  = blockIdx.x * 8 + (threadIdx.x >> 5);
    float* p = out + (size_t)row * HID + lane * 16;

    float4 v[4];
#pragma unroll
    for (int q = 0; q < 4; ++q) v[q] = ((const float4*)p)[q];

    float s = 0.f, ss = 0.f;
#pragma unroll
    for (int q = 0; q < 4; ++q) {
        s  += v[q].x + v[q].y + v[q].z + v[q].w;
        ss += v[q].x * v[q].x + v[q].y * v[q].y + v[q].z * v[q].z + v[q].w * v[q].w;
    }
#pragma unroll
    for (int off = 16; off > 0; off >>= 1) {
        s  += __shfl_xor(s,  off, 32);
        ss += __shfl_xor(ss, off, 32);
    }
    const float mu  = s * (1.0f / HID);
    const float var = ss * (1.0f / HID) - mu * mu;
    const float rs  = rsqrtf(var + LN_EPS);

    const float4* wv = (const float4*)(ln_w + lane * 16);
    const float4* bv = (const float4*)(ln_b + lane * 16);
#pragma unroll
    for (int q = 0; q < 4; ++q) {
        float4 w = wv[q], b = bv[q], o;
        o.x = (v[q].x - mu) * rs * w.x + b.x;
        o.y = (v[q].y - mu) * rs * w.y + b.y;
        o.z = (v[q].z - mu) * rs * w.z + b.z;
        o.w = (v[q].w - mu) * rs * w.w + b.w;
        ((float4*)p)[q] = o;
    }
}

// ---------------------------------------------------------------- launch

extern "C" void kernel_launch(void* const* d_in, const int* in_sizes, int n_in,
                              void* d_out, int out_size, void* d_ws, size_t ws_size,
                              hipStream_t stream) {
    (void)in_sizes; (void)n_in; (void)out_size; (void)ws_size;
    const float* x     = (const float*)d_in[0];  // (T*N, IN)
    const float* hxs   = (const float*)d_in[1];  // (N, H)
    const float* masks = (const float*)d_in[2];  // (T*N, 1)
    const float* Wih   = (const float*)d_in[3];  // (3H, IN)
    const float* Whh   = (const float*)d_in[4];  // (3H, H)
    const float* bih   = (const float*)d_in[5];
    const float* bhh   = (const float*)d_in[6];
    const float* lnw   = (const float*)d_in[7];
    const float* lnb   = (const float*)d_in[8];

    float* out    = (float*)d_out;                          // (T*N, H) normed
    float* h_last = out + (size_t)T_STEPS * N_ENV * HID;    // (N, H) un-normed tail

    __bf16* Wih_bf = (__bf16*)d_ws;
    __bf16* Whh_bf = Wih_bf + (size_t)H3 * IN_DIMC;

    cvt_weights_kernel<<<(H3 * IN_DIMC + 255) / 256, 256, 0, stream>>>(
        Wih, Whh, Wih_bf, Whh_bf);

    for (int t = 0; t < T_STEPS; ++t) {
        const float* x_t = x + (size_t)t * N_ENV * IN_DIMC;
        const float* hp  = (t == 0) ? hxs : (out + (size_t)(t - 1) * N_ENV * HID);
        const float* mt  = masks + (size_t)t * N_ENV;
        float* ho        = out + (size_t)t * N_ENV * HID;
        gru_step_kernel<<<128, 128, 0, stream>>>(
            x_t, hp, mt, Wih_bf, Whh_bf, bih, bhh, ho, h_last,
            (t == T_STEPS - 1) ? 1 : 0);
    }

    ln_kernel<<<(T_STEPS * N_ENV) / 8, 256, 0, stream>>>(out, lnw, lnb);
}
